// MultiHeadAttention_77524159693112
// MI455X (gfx1250) — matrix operand order repaired and checked by reading, not compiled, not run on hardware
//
#include <hip/hip_runtime.h>
#include <hip/hip_bf16.h>

#define BB    8
#define KSEQ  1024
#define DD    512
#define HH    8
#define DHEAD 64

typedef _Float16 v16h __attribute__((ext_vector_type(16)));
typedef _Float16 v8h  __attribute__((ext_vector_type(8)));
typedef float    v8f  __attribute__((ext_vector_type(8)));
typedef float    v4f  __attribute__((ext_vector_type(4)));

static __device__ inline v16h cat16(v8h lo, v8h hi) {
  return __builtin_shufflevector(lo, hi, 0,1,2,3,4,5,6,7,8,9,10,11,12,13,14,15);
}

static __device__ inline v8h cvt8(const float* p) {
  v4f a = *(const v4f*)p;
  v4f b = *(const v4f*)(p + 4);
  v8h r;
  #pragma unroll
  for (int i = 0; i < 4; i++) { r[i] = (_Float16)a[i]; r[4+i] = (_Float16)b[i]; }
  return r;
}

// A-matrix 16x32 f16 fragment (ISA 7.12.2): lane row = lane&15, g = lane>>4.
// halfs 0..7 -> K = g*8+0..7 ; halfs 8..15 -> K = 16+g*8+0..7.
static __device__ inline v16h fragA_f16(const _Float16* p, int g) {
  v8h lo = *(const v8h*)(p + g * 8);
  v8h hi = *(const v8h*)(p + 16 + g * 8);
  return cat16(lo, hi);
}
static __device__ inline v16h fragA_f32(const float* p, int g) {
  v8h lo = cvt8(p + g * 8);
  v8h hi = cvt8(p + 16 + g * 8);
  return cat16(lo, hi);
}
// B-matrix 32x16 f16 fragment: lane col = lane&15; same K order as A:
// halfs 0..7 -> K = g*8+0..7 ; halfs 8..15 -> K = 16+g*8+0..7.
static __device__ inline v16h fragB_f16(const _Float16* p, int g) {
  v8h lo = *(const v8h*)(p + g * 8);
  v8h hi = *(const v8h*)(p + 16 + g * 8);
  return cat16(lo, hi);
}

#define WMMA_F16(A, B, C) \
  __builtin_amdgcn_wmma_f32_16x16x32_f16(false, (A), false, (B), (short)0, (C), false, false)

// --- gfx1250 async global->LDS copy (ASYNCcnt-tracked, ISA §10/§15.18) -----
static __device__ inline unsigned lds_off(const void* p) {
  // flat shared address: addr[31:0] is the wave-relative LDS offset
  return (unsigned)(unsigned long long)p;
}
static __device__ inline void async_b128(unsigned lds, const void* gptr) {
  asm volatile("global_load_async_to_lds_b128 %0, %1, off"
               :: "v"(lds), "v"((unsigned long long)gptr) : "memory");
}
// ASYNCcnt decrements in order -> waiting <=N means "all but newest N done".
#define WAIT_ASYNC(n) asm volatile("s_wait_asynccnt " #n ::: "memory")

// ---------------------------------------------------------------------------
// Kernel 0: W[DD][DD] f32 row-major -> Wt[n][k] f16 (transposed)
// ---------------------------------------------------------------------------
__global__ void wt_convert_kernel(const float* __restrict__ W, _Float16* __restrict__ Wt) {
  int o  = blockIdx.x * blockDim.x + threadIdx.x;   // n*DD + k
  int n  = o >> 9;
  int kd = o & 511;
  Wt[o] = (_Float16)W[(size_t)kd * DD + n];
}

// stage one 64-col x 32-k B tile into LDS (rows padded to 40 halfs)
static __device__ inline void stage_b(_Float16* buf, const _Float16* Wt,
                                      int colBase, int kk, int tid) {
  #pragma unroll
  for (int s = 0; s < 2; s++) {
    int c = tid + s * 128;
    int n = c >> 2, o = c & 3;
    async_b128(lds_off(&buf[n * 40 + o * 8]),
               Wt + (size_t)(colBase + n) * DD + kk + o * 8);
  }
}

// ---------------------------------------------------------------------------
// Kernel 1: fused QKV projection. 128-thread block = 4 waves; block computes a
// 128-row x 64-col tile (one head's columns). Double-buffered async B staging:
// copy for step i+1 overlaps WMMAs of step i.
// ---------------------------------------------------------------------------
__global__ __launch_bounds__(128) void qkv_gemm_kernel(
    const float* __restrict__ Aq, const float* __restrict__ Ak, const float* __restrict__ Av,
    const _Float16* __restrict__ Wtq, const _Float16* __restrict__ Wtk, const _Float16* __restrict__ Wtv,
    const float* __restrict__ bq, const float* __restrict__ bk, const float* __restrict__ bv,
    _Float16* __restrict__ Qh, _Float16* __restrict__ Kh, _Float16* __restrict__ Vt)
{
  __shared__ __align__(16) _Float16 bsh[2][64 * 40];

  const int tid  = threadIdx.x;
  const int wave = tid >> 5;
  const int lane = tid & 31;
  const int g = lane >> 4, ln = lane & 15;
  const int rowBase = blockIdx.x * 128 + wave * 32;
  const int h = blockIdx.y;            // colBase = h*64
  const int which = blockIdx.z;

  const float*    A    = (which == 0) ? Aq  : (which == 1) ? Ak  : Av;
  const _Float16* Wt   = (which == 0) ? Wtq : (which == 1) ? Wtk : Wtv;
  const float*    bias = (which == 0) ? bq  : (which == 1) ? bk  : bv;

  const float* arow0 = A + (size_t)(rowBase + ln) * DD;
  const float* arow1 = arow0 + (size_t)16 * DD;

  v8f acc[2][4] = {};
  stage_b(bsh[0], Wt, h * 64, 0, tid);
  for (int kk = 0; kk < DD; kk += 32) {
    const int cur = (kk >> 5) & 1;
    __builtin_prefetch(arow0 + kk + 32, 0, 1);
    __builtin_prefetch(arow1 + kk + 32, 0, 1);
    v16h a0 = fragA_f32(arow0 + kk, g);
    v16h a1 = fragA_f32(arow1 + kk, g);
    if (kk + 32 < DD) {
      stage_b(bsh[cur ^ 1], Wt, h * 64, kk + 32, tid);
      WAIT_ASYNC(2);         // previous stage complete, next still in flight
    } else {
      WAIT_ASYNC(0);
    }
    __syncthreads();
    #pragma unroll
    for (int ch = 0; ch < 4; ch++) {
      v16h b = fragB_f16(&bsh[cur][(ch * 16 + ln) * 40], g);
      acc[0][ch] = WMMA_F16(a0, b, acc[0][ch]);
      acc[1][ch] = WMMA_F16(a1, b, acc[1][ch]);
    }
  }

  #pragma unroll
  for (int half = 0; half < 2; half++) {
    #pragma unroll
    for (int ch = 0; ch < 4; ch++) {
      int   d    = ch * 16 + ln;
      float bval = bias[h * 64 + d];
      #pragma unroll
      for (int r = 0; r < 8; r++) {
        int m   = rowBase + half * 16 + r + 8 * g;   // global token row
        int bIx = m >> 10, tok = m & 1023;
        float val = acc[half][ch][r] + bval;
        if (which == 2) {
          Vt[(((size_t)(bIx * HH + h)) * 64 + d) * KSEQ + tok] = (_Float16)val;
        } else {
          _Float16* dst = (which == 0) ? Qh : Kh;
          dst[(((size_t)(bIx * HH + h)) * KSEQ + tok) * 64 + d] = (_Float16)val;
        }
      }
    }
  }
}

// stage K (32 keys x 64 d, rows padded 72) + V (64 d x 32 keys, rows padded 40)
static __device__ inline void stage_kv(_Float16* kbuf, _Float16* vbuf,
                                       const _Float16* Kbase, const _Float16* Vbase,
                                       int j, int tid) {
  #pragma unroll
  for (int s = 0; s < 2; s++) {
    int c = tid + s * 128;
    int key = c >> 3, o = c & 7;
    async_b128(lds_off(&kbuf[key * 72 + o * 8]),
               Kbase + (size_t)(j + key) * 64 + o * 8);
    int d = c >> 2, ov = c & 3;
    async_b128(lds_off(&vbuf[d * 40 + ov * 8]),
               Vbase + (size_t)d * KSEQ + j + ov * 8);
  }
}

// ---------------------------------------------------------------------------
// Kernel 2: flash attention. 128-thread block = 4 waves x 16 queries of one
// (b,h). Double-buffered async K/V staging shared by all waves; per-wave
// online softmax with P re-laid out via padded per-wave LDS rows.
// ---------------------------------------------------------------------------
__global__ __launch_bounds__(128) void attn_kernel(
    const _Float16* __restrict__ Qh, const _Float16* __restrict__ Kh,
    const _Float16* __restrict__ Vt, _Float16* __restrict__ Ctx)
{
  __shared__ __align__(16) _Float16 ksh[2][32 * 72];
  __shared__ __align__(16) _Float16 vsh[2][64 * 40];
  __shared__ __align__(16) _Float16 psh[4 * 16 * 40];

  const int tid  = threadIdx.x;
  const int wave = tid >> 5;
  const int lane = tid & 31;
  const int g = lane >> 4, ln = lane & 15;
  const int qBase = blockIdx.x * 64 + wave * 16;
  const int h = blockIdx.y, b = blockIdx.z;

  const size_t headQK = ((size_t)(b * HH + h)) * KSEQ * 64;   // [tok][64]
  const size_t headV  = ((size_t)(b * HH + h)) * 64 * KSEQ;   // [d][tok]
  const _Float16* Kbase = Kh + headQK;
  const _Float16* Vbase = Vt + headV;

  const _Float16* qrow = Qh + headQK + (size_t)(qBase + ln) * 64;
  const v16h aq0 = fragA_f16(qrow + 0,  g);   // d = 0..31
  const v16h aq1 = fragA_f16(qrow + 32, g);   // d = 32..63

  _Float16* pb = &psh[wave * 16 * 40];

  v8f cacc[4] = {};
  float rm[8], rl[8];
  #pragma unroll
  for (int r = 0; r < 8; r++) { rm[r] = -1e30f; rl[r] = 0.0f; }

  const float scale = 0.03125f;   // 1/sqrt(KSEQ)

  stage_kv(ksh[0], vsh[0], Kbase, Vbase, 0, tid);
  for (int j = 0; j < KSEQ; j += 32) {
    const int cur = (j >> 5) & 1;
    if (j + 32 < KSEQ) {
      stage_kv(ksh[cur ^ 1], vsh[cur ^ 1], Kbase, Vbase, j + 32, tid);
      WAIT_ASYNC(4);         // previous K/V stage complete, next in flight
    } else {
      WAIT_ASYNC(0);
    }
    __syncthreads();

    const _Float16* kt = ksh[cur];
    const _Float16* vt = vsh[cur];

    // S0: keys j+ln (cols 0..15), S1: keys j+16+ln
    v8f s0 = {}, s1 = {};
    s0 = WMMA_F16(aq0, fragB_f16(&kt[ln * 72 + 0],  g), s0);
    s0 = WMMA_F16(aq1, fragB_f16(&kt[ln * 72 + 32], g), s0);
    s1 = WMMA_F16(aq0, fragB_f16(&kt[(16 + ln) * 72 + 0],  g), s1);
    s1 = WMMA_F16(aq1, fragB_f16(&kt[(16 + ln) * 72 + 32], g), s1);

    float cf[8];
    #pragma unroll
    for (int r = 0; r < 8; r++) {
      float x0 = s0[r] * scale, x1 = s1[r] * scale;
      float t = fmaxf(x0, x1);
      #pragma unroll
      for (int msk = 1; msk < 16; msk <<= 1) t = fmaxf(t, __shfl_xor(t, msk, 32));
      float nm = fmaxf(rm[r], t);
      float c  = __expf(rm[r] - nm);
      rm[r] = nm;
      float p0 = __expf(x0 - nm), p1 = __expf(x1 - nm);
      float ps = p0 + p1;
      #pragma unroll
      for (int msk = 1; msk < 16; msk <<= 1) ps += __shfl_xor(ps, msk, 32);
      rl[r] = rl[r] * c + ps;
      cf[r] = c;
      pb[(r + 8 * g) * 40 + ln]      = (_Float16)p0;   // row m, key col ln
      pb[(r + 8 * g) * 40 + 16 + ln] = (_Float16)p1;   // row m, key col 16+ln
    }
    #pragma unroll
    for (int ch = 0; ch < 4; ch++)
      #pragma unroll
      for (int r = 0; r < 8; r++) cacc[ch][r] *= cf[r];

    // P (16x32) as A fragment; same-wave LDS is in-order
    v16h ap = fragA_f16(&pb[ln * 40], g);
    #pragma unroll
    for (int ch = 0; ch < 4; ch++) {
      v16h bv = fragB_f16(&vt[(ch * 16 + ln) * 40], g);
      cacc[ch] = WMMA_F16(ap, bv, cacc[ch]);
    }
  }

  #pragma unroll
  for (int r = 0; r < 8; r++) {
    float inv = 1.0f / rl[r];
    int m = qBase + r + 8 * g;
    _Float16* crow = Ctx + ((size_t)(b * KSEQ + m)) * DD + h * 64;
    #pragma unroll
    for (int ch = 0; ch < 4; ch++)
      crow[ch * 16 + ln] = (_Float16)(cacc[ch][r] * inv);
  }
}

// ---------------------------------------------------------------------------
// Kernel 3: output projection. Same scheme, f32 result to d_out.
// ---------------------------------------------------------------------------
__global__ __launch_bounds__(128) void out_gemm_kernel(
    const _Float16* __restrict__ Ctx, const _Float16* __restrict__ Wto,
    const float* __restrict__ bo, float* __restrict__ Out)
{
  __shared__ __align__(16) _Float16 bsh[2][64 * 40];

  const int tid  = threadIdx.x;
  const int wave = tid >> 5;
  const int lane = tid & 31;
  const int g = lane >> 4, ln = lane & 15;
  const int rowBase = blockIdx.x * 128 + wave * 32;
  const int colBase = blockIdx.y * 64;

  const _Float16* arow0 = Ctx + (size_t)(rowBase + ln) * DD;
  const _Float16* arow1 = arow0 + (size_t)16 * DD;

  v8f acc[2][4] = {};
  stage_b(bsh[0], Wto, colBase, 0, tid);
  for (int kk = 0; kk < DD; kk += 32) {
    const int cur = (kk >> 5) & 1;
    __builtin_prefetch(arow0 + kk + 32, 0, 1);
    __builtin_prefetch(arow1 + kk + 32, 0, 1);
    v16h a0 = fragA_f16(arow0 + kk, g);
    v16h a1 = fragA_f16(arow1 + kk, g);
    if (kk + 32 < DD) {
      stage_b(bsh[cur ^ 1], Wto, colBase, kk + 32, tid);
      WAIT_ASYNC(2);
    } else {
      WAIT_ASYNC(0);
    }
    __syncthreads();
    #pragma unroll
    for (int ch = 0; ch < 4; ch++) {
      v16h bfr = fragB_f16(&bsh[cur][(ch * 16 + ln) * 40], g);
      acc[0][ch] = WMMA_F16(a0, bfr, acc[0][ch]);
      acc[1][ch] = WMMA_F16(a1, bfr, acc[1][ch]);
    }
  }

  #pragma unroll
  for (int half = 0; half < 2; half++) {
    #pragma unroll
    for (int ch = 0; ch < 4; ch++) {
      int n = colBase + ch * 16 + ln;
      float bval = bo[n];
      #pragma unroll
      for (int r = 0; r < 8; r++) {
        int m = rowBase + half * 16 + r + 8 * g;
        Out[(size_t)m * DD + n] = acc[half][ch][r] + bval;
      }
    }
  }
}

// ---------------------------------------------------------------------------
extern "C" void kernel_launch(void* const* d_in, const int* in_sizes, int n_in,
                              void* d_out, int out_size, void* d_ws, size_t ws_size,
                              hipStream_t stream) {
  const float* query = (const float*)d_in[0];
  const float* key   = (const float*)d_in[1];
  const float* value = (const float*)d_in[2];
  const float* Wq    = (const float*)d_in[3];
  const float* bq    = (const float*)d_in[4];
  const float* Wk    = (const float*)d_in[5];
  const float* bk    = (const float*)d_in[6];
  const float* Wv    = (const float*)d_in[7];
  const float* bv    = (const float*)d_in[8];
  const float* Wo    = (const float*)d_in[9];
  const float* bo    = (const float*)d_in[10];
  float* out = (float*)d_out;

  char* ws = (char*)d_ws;
  const size_t WSZ  = (size_t)DD * DD * sizeof(_Float16);             // 512 KB
  const size_t QKSZ = (size_t)BB * HH * KSEQ * 64 * sizeof(_Float16); // 8 MB
  _Float16* Wtq = (_Float16*)(ws + 0 * WSZ);
  _Float16* Wtk = (_Float16*)(ws + 1 * WSZ);
  _Float16* Wtv = (_Float16*)(ws + 2 * WSZ);
  _Float16* Wto = (_Float16*)(ws + 3 * WSZ);
  _Float16* Qh  = (_Float16*)(ws + 4 * WSZ);
  _Float16* Kh  = (_Float16*)(ws + 4 * WSZ + 1 * QKSZ);
  _Float16* Vt  = (_Float16*)(ws + 4 * WSZ + 2 * QKSZ);
  _Float16* Ctx = (_Float16*)(ws + 4 * WSZ + 3 * QKSZ);

  wt_convert_kernel<<<DD * DD / 256, 256, 0, stream>>>(Wq, Wtq);
  wt_convert_kernel<<<DD * DD / 256, 256, 0, stream>>>(Wk, Wtk);
  wt_convert_kernel<<<DD * DD / 256, 256, 0, stream>>>(Wv, Wtv);
  wt_convert_kernel<<<DD * DD / 256, 256, 0, stream>>>(Wo, Wto);

  qkv_gemm_kernel<<<dim3(BB * KSEQ / 128, HH, 3), 128, 0, stream>>>(
      query, key, value, Wtq, Wtk, Wtv, bq, bk, bv, Qh, Kh, Vt);

  attn_kernel<<<dim3(KSEQ / 64, HH, BB), 128, 0, stream>>>(Qh, Kh, Vt, Ctx);

  out_gemm_kernel<<<dim3(BB * KSEQ / 128, DD / 64), 128, 0, stream>>>(Ctx, Wto, bo, out);
}
